// cnn2D_13975823582218
// MI455X (gfx1250) — matrix operand order repaired and checked by reading, not compiled, not run on hardware
//
#include <hip/hip_runtime.h>

#define B_      32
#define CIN     128
#define H_      64
#define W_      64
#define COUT    128
#define NK      4
#define HID     32
#define KDIM    1152          // CIN*3*3
#define SPATIAL 4096          // H*W
#define TEMP    34.0f

typedef __attribute__((ext_vector_type(16))) __bf16        v16bf;
typedef __attribute__((ext_vector_type(8)))  float         v8f;
typedef __attribute__((ext_vector_type(8)))  unsigned short ushort8x;

union Frag { v16bf v; ushort8x h[2]; };

__device__ __forceinline__ unsigned short f2bf(float f) {
    unsigned u = __builtin_bit_cast(unsigned, f);
    u += 0x7FFFu + ((u >> 16) & 1u);          // round-to-nearest-even
    return (unsigned short)(u >> 16);
}

// ---------------------------------------------------------------------------
// Kernel 1: context[b,c] = mean over H,W of x[b,c,:,:]
// grid = B*CIN blocks, 256 threads
// ---------------------------------------------------------------------------
__global__ void ctx_kernel(const float* __restrict__ x, float* __restrict__ ctx) {
    __shared__ float red[256];
    const int bc  = blockIdx.x;
    const int tid = threadIdx.x;
    const float* p = x + (size_t)bc * SPATIAL;
    float s = 0.f;
    #pragma unroll 4
    for (int i = tid; i < SPATIAL; i += 256) s += p[i];
    red[tid] = s;
    __syncthreads();
    for (int off = 128; off > 0; off >>= 1) {
        if (tid < off) red[tid] += red[tid + off];
        __syncthreads();
    }
    if (tid == 0) ctx[bc] = red[0] * (1.0f / (float)SPATIAL);
}

// ---------------------------------------------------------------------------
// Kernel 2: attention MLP + softmax + aggregated bias. One block, 256 thr.
// ---------------------------------------------------------------------------
__global__ void attn_kernel(const float* __restrict__ ctx,
                            const float* __restrict__ fc1,   // [HID][CIN]
                            const float* __restrict__ fc2,   // [NK][HID]
                            const float* __restrict__ fc2b,  // [NK]
                            const float* __restrict__ bias,  // [NK][COUT]
                            float* __restrict__ att,         // [B][NK]
                            float* __restrict__ aggb) {      // [B][COUT]
    __shared__ float s_att[B_ * NK];
    const int tid = threadIdx.x;
    if (tid < B_) {
        const int b = tid;
        float hid[HID];
        #pragma unroll
        for (int h = 0; h < HID; ++h) {
            float s = 0.f;
            for (int c = 0; c < CIN; ++c) s += ctx[b * CIN + c] * fc1[h * CIN + c];
            hid[h] = s > 0.f ? s : 0.f;
        }
        float lg[NK], mx = -1e30f;
        #pragma unroll
        for (int j = 0; j < NK; ++j) {
            float s = fc2b[j];
            #pragma unroll
            for (int h = 0; h < HID; ++h) s += hid[h] * fc2[j * HID + h];
            lg[j] = s * (1.0f / TEMP);
            mx = fmaxf(mx, lg[j]);
        }
        float den = 0.f;
        #pragma unroll
        for (int j = 0; j < NK; ++j) { lg[j] = __expf(lg[j] - mx); den += lg[j]; }
        const float inv = 1.0f / den;
        #pragma unroll
        for (int j = 0; j < NK; ++j) {
            const float a = lg[j] * inv;
            s_att[b * NK + j] = a;
            att[b * NK + j]   = a;
        }
    }
    __syncthreads();
    // agg_b[b][co] = sum_j att[b][j] * bias[j][co]   (B*COUT = 4096 outputs)
    for (int idx = tid; idx < B_ * COUT; idx += 256) {
        const int b  = idx >> 7;
        const int co = idx & 127;
        float s = 0.f;
        #pragma unroll
        for (int j = 0; j < NK; ++j) s += s_att[b * NK + j] * bias[j * COUT + co];
        aggb[idx] = s;
    }
}

// ---------------------------------------------------------------------------
// Kernel 3: aggW_bf16[b][cout][k] = sum_j att[b][j] * W[j][cout][k], k=ci*9+kh*3+kw
// grid = B*COUT*KDIM/256 blocks
// ---------------------------------------------------------------------------
__global__ void aggw_kernel(const float* __restrict__ weight, // [NK][COUT*KDIM]
                            const float* __restrict__ att,    // [B][NK]
                            unsigned short* __restrict__ aggW) {
    const int t = blockIdx.x * 256 + threadIdx.x;
    const int per = COUT * KDIM;                 // 147456
    const int b = t / per;
    const int r = t - b * per;
    const float a0 = att[b * NK + 0], a1 = att[b * NK + 1];
    const float a2 = att[b * NK + 2], a3 = att[b * NK + 3];
    const float v = a0 * weight[r] + a1 * weight[per + r] +
                    a2 * weight[2 * per + r] + a3 * weight[3 * per + r];
    aggW[t] = f2bf(v);
}

// ---------------------------------------------------------------------------
// Kernel 4: implicit-GEMM conv via WMMA bf16.
// Per sample b: C[128,4096] = A[128,1152] x im2col(x_b)[1152,4096]
// Tile: BM=128 x BN=128 x BK=32, 256 threads = 8 waves (2M x 4N),
// each wave: 4x2 fragments of v_wmma_f32_16x16x32_bf16.
// grid = (SPATIAL/BN, B)
// ---------------------------------------------------------------------------
#define BN   128
#define BK   32
#define LSTR 40     // padded LDS row stride (bf16): 80B, 16B-aligned, conflict-free frags

__global__ __launch_bounds__(256) void conv_kernel(
        const float* __restrict__ x,            // [B][CIN][H][W] fp32
        const unsigned short* __restrict__ aggW,// [B][COUT][KDIM] bf16
        const float* __restrict__ aggb,         // [B][COUT]
        float* __restrict__ out) {              // [B][COUT][H][W]
    const int tid   = threadIdx.x;
    const int lane  = tid & 31;
    const int wave  = tid >> 5;
    const int waveM = wave >> 2;                // 0..1  (64 rows each)
    const int waveN = wave & 3;                 // 0..3  (32 cols each)
    const int nblk  = blockIdx.x;               // 0..31
    const int b     = blockIdx.y;               // 0..31

    __shared__ alignas(16) unsigned short Atile[COUT * LSTR];
    __shared__ alignas(16) unsigned short Btile[BN * LSTR];

    v8f c[4][2];
    const v8f vzero = {0.f, 0.f, 0.f, 0.f, 0.f, 0.f, 0.f, 0.f};
    #pragma unroll
    for (int fi = 0; fi < 4; ++fi)
        #pragma unroll
        for (int fj = 0; fj < 2; ++fj) c[fi][fj] = vzero;

    const unsigned short* Ag = aggW + (size_t)b * (COUT * KDIM);
    const float*          xb = x    + (size_t)b * (CIN * SPATIAL);
    const int n0g = nblk * BN;

    // Fragment lane geometry (ISA 7.12.2, 16-bit A 16x32 / B 32x16, f32 C 16x16)
    const int amRow = lane & 15;
    const int aKoff = (lane < 16) ? 0 : 8;      // h[0]@aKoff, h[1]@aKoff+16
    const int bNcol = lane & 15;
    const int bKoff = (lane < 16) ? 0 : 8;      // h[0]@bKoff, h[1]@bKoff+16

    // B-stage geometry: threads 0-127 -> n=tid, kk even; 128-255 -> kk odd
    const int stK = tid >> 7;
    const int stN = tid & 127;
    const int ng  = n0g + stN;
    const int oh  = ng >> 6;
    const int ow  = ng & 63;

    for (int ks = 0; ks < KDIM / BK; ++ks) {    // 36 steps
        __syncthreads();
        // ---- stage A: dense 128x32 bf16 block (32B per thread, b128 loads)
        {
            const int m = tid >> 1, half = tid & 1;
            const ushort8x* src =
                (const ushort8x*)(Ag + m * KDIM + ks * BK + half * 16);
            ushort8x v0 = src[0];
            ushort8x v1 = src[1];
            *(ushort8x*)&Atile[m * LSTR + half * 16]     = v0;
            *(ushort8x*)&Atile[m * LSTR + half * 16 + 8] = v1;
        }
        // ---- stage B: im2col gather from fp32 x, convert to bf16, store [N][K]
        #pragma unroll
        for (int i = 0; i < 16; ++i) {
            const int kk = stK + i * 2;
            const int kg = ks * BK + kk;
            const int ci = kg / 9;
            const int rr = kg - ci * 9;
            const int kh = rr / 3;
            const int kw = rr - kh * 3;
            const int ih = oh + kh - 1;
            const int iw = ow + kw - 1;
            float v = 0.f;
            if ((unsigned)ih < (unsigned)H_ && (unsigned)iw < (unsigned)W_)
                v = xb[ci * SPATIAL + ih * W_ + iw];
            Btile[stN * LSTR + kk] = f2bf(v);
        }
        __syncthreads();

        // ---- load fragments (conflict-free ds_load_b128) and WMMA
        Frag a[4], bm[2];
        #pragma unroll
        for (int fi = 0; fi < 4; ++fi) {
            const int m = waveM * 64 + fi * 16 + amRow;
            a[fi].h[0] = *(const ushort8x*)&Atile[m * LSTR + aKoff];
            a[fi].h[1] = *(const ushort8x*)&Atile[m * LSTR + aKoff + 16];
        }
        #pragma unroll
        for (int fj = 0; fj < 2; ++fj) {
            const int n = waveN * 32 + fj * 16 + bNcol;
            bm[fj].h[0] = *(const ushort8x*)&Btile[n * LSTR + bKoff];
            bm[fj].h[1] = *(const ushort8x*)&Btile[n * LSTR + bKoff + 16];
        }
        #pragma unroll
        for (int fi = 0; fi < 4; ++fi)
            #pragma unroll
            for (int fj = 0; fj < 2; ++fj)
                c[fi][fj] = __builtin_amdgcn_wmma_f32_16x16x32_bf16(
                    false, a[fi].v, false, bm[fj].v,
                    (short)0, c[fi][fj], false, false);
    }

    // ---- epilogue: D layout M = e + 8*(lane>=16), N = lane&15; add bias
    float* outb = out + (size_t)b * (COUT * SPATIAL);
    const int mBase = waveM * 64 + ((lane >> 4) << 3);
    const int nBase = n0g + waveN * 32 + (lane & 15);
    #pragma unroll
    for (int fi = 0; fi < 4; ++fi) {
        #pragma unroll
        for (int fj = 0; fj < 2; ++fj) {
            const int nc = nBase + fj * 16;
            #pragma unroll
            for (int e = 0; e < 8; ++e) {
                const int m = mBase + fi * 16 + e;
                outb[m * SPATIAL + nc] = c[fi][fj][e] + aggb[b * COUT + m];
            }
        }
    }
}

// ---------------------------------------------------------------------------
extern "C" void kernel_launch(void* const* d_in, const int* in_sizes, int n_in,
                              void* d_out, int out_size, void* d_ws, size_t ws_size,
                              hipStream_t stream) {
    const float* x     = (const float*)d_in[0];
    const float* fc1_w = (const float*)d_in[1];
    const float* fc2_w = (const float*)d_in[2];
    const float* fc2_b = (const float*)d_in[3];
    const float* weight= (const float*)d_in[4];
    const float* bias  = (const float*)d_in[5];
    float* out = (float*)d_out;

    // Workspace layout
    float* ctx  = (float*)d_ws;                 // B*CIN          = 4096 f32
    float* att  = ctx + B_ * CIN;               // B*NK           = 128  f32
    float* aggb = att + B_ * NK;                // B*COUT         = 4096 f32
    unsigned short* aggW = (unsigned short*)(aggb + B_ * COUT); // B*COUT*KDIM bf16 (~9.4MB)

    ctx_kernel<<<dim3(B_ * CIN), dim3(256), 0, stream>>>(x, ctx);
    attn_kernel<<<dim3(1), dim3(256), 0, stream>>>(ctx, fc1_w, fc2_w, fc2_b,
                                                   bias, att, aggb);
    aggw_kernel<<<dim3((B_ * COUT * KDIM) / 256), dim3(256), 0, stream>>>(
        weight, att, aggW);
    conv_kernel<<<dim3(SPATIAL / BN, B_), dim3(256), 0, stream>>>(
        x, aggW, aggb, out);
}
